// TemporalEvolutionTrust_76836964926330
// MI455X (gfx1250) — matrix operand order repaired and checked by reading, not compiled, not run on hardware
//
#include <hip/hip_runtime.h>
#include <math.h>

typedef __attribute__((ext_vector_type(16))) __bf16 v16bf;
typedef __attribute__((ext_vector_type(8)))  __bf16 v8bf;
typedef __attribute__((ext_vector_type(8)))  float  v8f;
typedef __attribute__((ext_vector_type(4)))  float  v4f;

#define DIM 256
#define NH 4
#define HD 64
#define WW 8
#define TE 16            // edges per block
#define ROWS 128         // TE * WW
#define LSTR 264         // padded bf16 row stride (528B, 16B aligned, breaks bank ≡0)
#define PSTR 260         // pooled f32 row stride
#define SCALE 0.125f     // HD^-0.5

// ---- fragment loaders -------------------------------------------------------

// A-matrix 16x32 bf16, contiguous k order used here (B matches): lanes 0-15 (row M=lane): K 0..15;
// lanes 16-31 (row M=lane-16): K 16..31.
__device__ __forceinline__ v16bf load_a_frag(const __bf16* row, int ks, int hi) {
    union { v16bf v; uint4 u[2]; } r;
    const int base = ks * 32 + (hi ? 16 : 0);
    r.u[0] = *(const uint4*)(row + base);
    r.u[1] = *(const uint4*)(row + base + 8);
    return r.v;
}

// B-matrix 32x16 bf16, same contiguous k order as A: lane n%16 holds column n; lanes 0-15: K 0..15,
// lanes 16-31: K 16..31 (contiguous in W[n][k] row-major storage).
__device__ __forceinline__ v16bf load_b_frag(const __bf16* Wb, int n, int kb) {
    union { v16bf v; uint4 u[2]; } r;
    const uint4* p = (const uint4*)(Wb + n * DIM + kb);
    r.u[0] = p[0];
    r.u[1] = p[1];
    return r.v;
}

// ---- weight pre-conversion (f32 -> bf16 in workspace) -----------------------

__global__ __launch_bounds__(256) void tet_convert_weights(
    const float* __restrict__ Wq, const float* __restrict__ Wk,
    const float* __restrict__ Wv,
    __bf16* __restrict__ Wqb, __bf16* __restrict__ Wkb, __bf16* __restrict__ Wvb)
{
    int i = blockIdx.x * 256 + threadIdx.x;   // exactly DIM*DIM threads
    Wqb[i] = (__bf16)Wq[i];
    Wkb[i] = (__bf16)Wk[i];
    Wvb[i] = (__bf16)Wv[i];
}

// ---- main fused kernel ------------------------------------------------------

__global__ __launch_bounds__(256) void tet_kernel(
    const float* __restrict__ h_i,   const float* __restrict__ evol_j,
    const float* __restrict__ evt,   const float* __restrict__ var_i,
    const float* __restrict__ var_j, const float* __restrict__ cur_t,
    const float* __restrict__ bq,    const float* __restrict__ bk,
    const float* __restrict__ bv,    const float* __restrict__ tdec,
    const float* __restrict__ Wu,    const float* __restrict__ bu,
    const float* __restrict__ Wo,    const float* __restrict__ bo,
    const __bf16* __restrict__ Wqb,  const __bf16* __restrict__ Wkb,
    const __bf16* __restrict__ Wvb,  float* __restrict__ out)
{
    __shared__ __bf16 sEvol[ROWS][LSTR];   // 67584 B
    __shared__ __bf16 sH[TE][LSTR];        //  8448 B
    __shared__ __bf16 sQ[TE][LSTR];        //  8448 B  (aliased as sRed after scores)
    __shared__ __bf16 sK[ROWS][LSTR];      // 67584 B  (aliased as pooled after scores)
    __shared__ float  sAttn[TE][NH][WW];   //  2048 B
    __shared__ float  sGate[TE][WW];       //   512 B
    __shared__ float  sTB[TE][WW];         //   512 B
    __shared__ float  sWu[DIM];            //  1024 B
    __shared__ float  sWo[DIM];            //  1024 B
    // total ≈ 153.5 KB static LDS

    const int t    = threadIdx.x;
    const int lane = t & 31;
    const int wave = t >> 5;
    const int lm   = lane & 15;
    const int hi   = lane >> 4;
    const long long eBase = (long long)blockIdx.x * TE;

    // ---------------- Phase 1: stream inputs, convert f32 -> bf16 in LDS ----
    {
        const float* src = evol_j + eBase * (WW * DIM);
        for (int i = t; i < (ROWS * DIM) / 4; i += 256) {
            v4f v = ((const v4f*)src)[i];
            int flat = i * 4, row = flat >> 8, col = flat & 255;
            union { __bf16 b[4]; uint2 u; } pk;
            pk.b[0] = (__bf16)v[0]; pk.b[1] = (__bf16)v[1];
            pk.b[2] = (__bf16)v[2]; pk.b[3] = (__bf16)v[3];
            *(uint2*)&sEvol[row][col] = pk.u;
        }
        const float* hs = h_i + eBase * DIM;
        for (int i = t; i < (TE * DIM) / 4; i += 256) {
            v4f v = ((const v4f*)hs)[i];
            int flat = i * 4, row = flat >> 8, col = flat & 255;
            union { __bf16 b[4]; uint2 u; } pk;
            pk.b[0] = (__bf16)v[0]; pk.b[1] = (__bf16)v[1];
            pk.b[2] = (__bf16)v[2]; pk.b[3] = (__bf16)v[3];
            *(uint2*)&sH[row][col] = pk.u;
        }
        if (t < DIM) { sWu[t] = Wu[t]; sWo[t] = Wo[t]; }
    }
    __syncthreads();

    // ---------------- Phase 2a: uncertainty gate + time-decay bias ----------
    if (t < ROWS) {
        int row = t, e = row >> 3, w = row & 7;
        float acc = 0.f;
        #pragma unroll 4
        for (int d8 = 0; d8 < DIM / 8; ++d8) {
            v8bf ev = *(const v8bf*)&sEvol[row][d8 * 8];
            #pragma unroll
            for (int u = 0; u < 8; ++u)
                acc += (float)ev[u] * sWu[d8 * 8 + u];
        }
        acc += bu[0];
        float g  = 1.f / (1.f + __expf(-acc));
        float dt = fmaxf(cur_t[eBase + e] - evt[(eBase + e) * WW + w], 0.f);
        sGate[e][w] = g;
        sTB[e][w]   = -fabsf(tdec[0]) * dt;
    }

    // ---------------- Phase 2b: Q = h_i @ Wq^T + bq (WMMA) ------------------
    {
        v16bf a[8];
        const __bf16* arow = &sH[lm][0];
        #pragma unroll
        for (int ks = 0; ks < 8; ++ks) a[ks] = load_a_frag(arow, ks, hi);
        #pragma unroll
        for (int j = 0; j < 2; ++j) {
            int nt = wave * 2 + j;
            v8f c = {};
            #pragma unroll
            for (int ks = 0; ks < 8; ++ks) {
                v16bf b = load_b_frag(Wqb, nt * 16 + lm, ks * 32 + hi * 16);
                c = __builtin_amdgcn_wmma_f32_16x16x32_bf16(
                        false, a[ks], false, b, (short)0, c, false, false);
            }
            int n = nt * 16 + lm;
            float bias = bq[n];
            #pragma unroll
            for (int v = 0; v < 8; ++v)
                sQ[hi * 8 + v][n] = (__bf16)(c[v] + bias);
        }
    }

    // ---------------- Phase 2c: K = evol @ Wk^T + bk (WMMA) -----------------
    {
        const int mt = wave;                       // 16-row M-tile
        v16bf a[8];
        const __bf16* arow = &sEvol[mt * 16 + lm][0];
        #pragma unroll
        for (int ks = 0; ks < 8; ++ks) a[ks] = load_a_frag(arow, ks, hi);
        for (int nt = 0; nt < 16; ++nt) {
            v8f c = {};
            #pragma unroll
            for (int ks = 0; ks < 8; ++ks) {
                v16bf b = load_b_frag(Wkb, nt * 16 + lm, ks * 32 + hi * 16);
                c = __builtin_amdgcn_wmma_f32_16x16x32_bf16(
                        false, a[ks], false, b, (short)0, c, false, false);
            }
            int n = nt * 16 + lm;
            float bias = bk[n];
            #pragma unroll
            for (int v = 0; v < 8; ++v)
                sK[mt * 16 + hi * 8 + v][n] = (__bf16)(c[v] + bias);
        }
    }
    __syncthreads();

    // ---------------- Phase 3: attention logits -----------------------------
    #pragma unroll
    for (int it = 0; it < 2; ++it) {
        int idx = t + it * 256;                    // 0..511 = (e, head, w)
        int e = idx >> 5, h = (idx >> 3) & 3, w = idx & 7;
        const __bf16* qrow = &sQ[e][h * HD];
        const __bf16* krow = &sK[e * WW + w][h * HD];
        float acc = 0.f;
        #pragma unroll
        for (int d8 = 0; d8 < HD / 8; ++d8) {
            v8bf qv = *(const v8bf*)(qrow + d8 * 8);
            v8bf kv = *(const v8bf*)(krow + d8 * 8);
            #pragma unroll
            for (int u = 0; u < 8; ++u)
                acc += (float)qv[u] * (float)kv[u];
        }
        sAttn[e][h][w] = (acc * SCALE + sTB[e][w]) * sGate[e][w];
    }
    __syncthreads();

    // ---------------- Phase 4: softmax over W -------------------------------
    if (t < TE * NH) {
        int e = t >> 2, h = t & 3;
        float m = -1e30f;
        #pragma unroll
        for (int w = 0; w < WW; ++w) m = fmaxf(m, sAttn[e][h][w]);
        float ex[WW], s = 0.f;
        #pragma unroll
        for (int w = 0; w < WW; ++w) { ex[w] = __expf(sAttn[e][h][w] - m); s += ex[w]; }
        float inv = 1.f / s;
        #pragma unroll
        for (int w = 0; w < WW; ++w) sAttn[e][h][w] = ex[w] * inv;
    }
    __syncthreads();

    // ---------------- Phase 5: V GEMM (WMMA) fused with attention pooling ---
    // sK is dead now; reuse its LDS for the pooled f32 tile [TE][PSTR].
    float* sPooled = (float*)sK;
    {
        const int mt = wave;
        v16bf a[8];
        const __bf16* arow = &sEvol[mt * 16 + lm][0];
        #pragma unroll
        for (int ks = 0; ks < 8; ++ks) a[ks] = load_a_frag(arow, ks, hi);
        const int eloc = mt * 2 + hi;              // this lane's edge
        for (int nt = 0; nt < 16; ++nt) {
            v8f c = {};
            #pragma unroll
            for (int ks = 0; ks < 8; ++ks) {
                v16bf b = load_b_frag(Wvb, nt * 16 + lm, ks * 32 + hi * 16);
                c = __builtin_amdgcn_wmma_f32_16x16x32_bf16(
                        false, a[ks], false, b, (short)0, c, false, false);
            }
            int n = nt * 16 + lm;
            int h = nt >> 2;                       // 16-dim tile lies in one head
            float bias = bv[n];
            float aw[WW];
            #pragma unroll
            for (int w = 0; w < WW; ++w) aw[w] = sAttn[eloc][h][w];
            float p = 0.f;
            #pragma unroll
            for (int v = 0; v < 8; ++v)            // C row v == window w of eloc
                p += aw[v] * (c[v] + bias);
            sPooled[eloc * PSTR + n] = p;
        }
    }
    __syncthreads();

    // ---------------- Phase 6: trust logit + confidences + sigmoid ----------
    float* sRed = (float*)sQ;                       // sQ dead; 3*16*16 floats
    {
        int e = t >> 4, c = t & 15;
        const float* vi = var_i + (eBase + e) * DIM + c * 16;
        const float* vj = var_j + (eBase + e) * DIM + c * 16;
        float so = 0.f, svi = 0.f, svj = 0.f;
        #pragma unroll
        for (int i = 0; i < 16; ++i) {
            so  += sPooled[e * PSTR + c * 16 + i] * sWo[c * 16 + i];
            svi += vi[i];
            svj += vj[i];
        }
        sRed[(0 * TE + e) * 16 + c] = so;
        sRed[(1 * TE + e) * 16 + c] = svi;
        sRed[(2 * TE + e) * 16 + c] = svj;
    }
    __syncthreads();
    if (t < TE) {
        float so = bo[0], svi = 0.f, svj = 0.f;
        #pragma unroll
        for (int c = 0; c < 16; ++c) {
            so  += sRed[(0 * TE + t) * 16 + c];
            svi += sRed[(1 * TE + t) * 16 + c];
            svj += sRed[(2 * TE + t) * 16 + c];
        }
        float ci = 1.f / (1.f + fmaxf(sqrtf(svi * (1.f / DIM)), 1e-6f));
        float cj = 1.f / (1.f + fmaxf(sqrtf(svj * (1.f / DIM)), 1e-6f));
        float z  = so * ci * cj;
        out[eBase + t] = 1.f / (1.f + __expf(-z));
    }
}

// ---- launcher ---------------------------------------------------------------

extern "C" void kernel_launch(void* const* d_in, const int* in_sizes, int n_in,
                              void* d_out, int out_size, void* d_ws, size_t ws_size,
                              hipStream_t stream) {
    const float* h_i  = (const float*)d_in[0];
    // d_in[1] = h_j (unused by reference output path)
    const float* evol = (const float*)d_in[2];
    const float* evt  = (const float*)d_in[3];
    const float* vari = (const float*)d_in[4];
    const float* varj = (const float*)d_in[5];
    const float* ct   = (const float*)d_in[6];
    const float* Wq   = (const float*)d_in[7];
    const float* bq   = (const float*)d_in[8];
    const float* Wk   = (const float*)d_in[9];
    const float* bk   = (const float*)d_in[10];
    const float* Wv   = (const float*)d_in[11];
    const float* bv   = (const float*)d_in[12];
    const float* td   = (const float*)d_in[13];
    const float* Wu   = (const float*)d_in[14];
    const float* bu   = (const float*)d_in[15];
    const float* Wo   = (const float*)d_in[16];
    const float* bo   = (const float*)d_in[17];
    float* out = (float*)d_out;

    __bf16* Wqb = (__bf16*)d_ws;
    __bf16* Wkb = Wqb + DIM * DIM;
    __bf16* Wvb = Wkb + DIM * DIM;

    tet_convert_weights<<<(DIM * DIM) / 256, 256, 0, stream>>>(Wq, Wk, Wv, Wqb, Wkb, Wvb);

    int nblk = out_size / TE;   // 100000 / 16 = 6250
    tet_kernel<<<nblk, 256, 0, stream>>>(h_i, evol, evt, vari, varj, ct,
                                         bq, bk, bv, td, Wu, bu, Wo, bo,
                                         Wqb, Wkb, Wvb, out);
    (void)in_sizes; (void)n_in; (void)ws_size;
}
